// DoubleLSTM_68496138437440
// MI455X (gfx1250) — matrix operand order repaired and checked by reading, not compiled, not run on hardware
//
#include <hip/hip_runtime.h>
#include <hip/hip_bf16.h>

typedef __bf16 bf16_t;
typedef __attribute__((ext_vector_type(16))) __bf16 v16bf;
typedef __attribute__((ext_vector_type(8)))  __bf16 v8bf;
typedef __attribute__((ext_vector_type(8)))  float  v8f;
typedef __attribute__((ext_vector_type(4)))  float  v4f;

#define B_  256
#define T_  2048
#define D_  32
#define H_  64
#define G_  256   /* 4H */

#define HS  72    /* h1 LDS row stride (bf16): 64 + pad (144B, 16B aligned) */
#define AS2 136   /* A2 row stride (bf16): K=128 + pad                      */
#define ZS  260   /* z row stride (f32): 256 + pad                          */

// Split barrier: only order LDS (DScnt) before signaling — do NOT drain
// LOADcnt, so the global x-prefetch issued this step stays in flight across
// the barrier and is waited only at next use (one full step later).
__device__ __forceinline__ void wg_barrier() {
  asm volatile("s_wait_dscnt 0\n\t"
               "s_barrier_signal -1\n\t"
               "s_barrier_wait -1" ::: "memory");
}

__device__ __forceinline__ float fsigmoid(float x) {
  // 1/(1+e^-x) with fast v_rcp_f32 (no IEEE div refinement chain)
  return __builtin_amdgcn_rcpf(1.0f + __expf(-x));
}
__device__ __forceinline__ float ftanh(float x) {
  float e = __expf(-2.0f * fabsf(x));             // e in (0,1] -> robust
  float r = __builtin_amdgcn_rcpf(1.0f + e);
  return copysignf(fmaf(-e, r, r), x);            // (1-e)/(1+e)
}

__global__ __launch_bounds__(512)
void lstm2_fused_kernel(const float* __restrict__ x,
                        const float* __restrict__ W1, const float* __restrict__ U1,
                        const float* __restrict__ b1,
                        const float* __restrict__ W2, const float* __restrict__ U2,
                        const float* __restrict__ b2,
                        const float* __restrict__ Wd, const float* __restrict__ bd,
                        float* __restrict__ out)
{
  __shared__ bf16_t sH1B[16 * HS];   // h1 (bf16) : A operand tiles kt=1,2 of GEMM1
  __shared__ bf16_t sA2 [16 * AS2];  // [h1' (K 0..63) | h2 (K 64..127)] (bf16)
  __shared__ float  sZ1 [16 * ZS];
  __shared__ float  sZ2 [16 * ZS];
  __shared__ float  sC1 [16 * H_];
  __shared__ float  sC2 [16 * H_];
  __shared__ float  sH2 [16 * H_];

  const int tid  = threadIdx.x;
  const int wave = tid >> 5;          // 0..15 : gate-column tile / head row
  const int lane = tid & 31;
  const int sub  = lane & 15;
  const int half = lane >> 4;
  const int nn   = wave * 16 + sub;   // gate column owned in B/C operands
  const int r0   = blockIdx.x * 16;   // batch-row base for this WG

  // ---- Preload B operands into registers in WMMA B layout -----------------
  // B (32x16 bf16): lane<16 -> N=lane, K = kt*32 + {0..7,16..23} ; lane>=16 -> K += 8
  v16bf b1r[3];   // [W1;U1] : K = 96
  #pragma unroll
  for (int kt = 0; kt < 3; ++kt) {
    v16bf b;
    #pragma unroll
    for (int j = 0; j < 16; ++j) {
      int kk = kt * 32 + half * 8 + j + (j >= 8 ? 8 : 0);
      float wv = (kt == 0) ? W1[kk * G_ + nn] : U1[(kk - 32) * G_ + nn];
      b[j] = (bf16_t)wv;
    }
    b1r[kt] = b;
  }
  v16bf b2r[4];   // [W2;U2] : K = 128
  #pragma unroll
  for (int kt = 0; kt < 4; ++kt) {
    v16bf b;
    #pragma unroll
    for (int j = 0; j < 16; ++j) {
      int kk = kt * 32 + half * 8 + j + (j >= 8 ? 8 : 0);
      float wv = (kt < 2) ? W2[kk * G_ + nn] : U2[(kk - 64) * G_ + nn];
      b[j] = (bf16_t)wv;
    }
    b2r[kt] = b;
  }

  const float bias1 = b1[nn];
  const float bias2 = b2[nn];
  const float wda   = Wd[lane];
  const float wdb   = Wd[lane + 32];
  const float bd0   = bd[0];

  // ---- Zero initial state (h1, h2, c1, c2) --------------------------------
  for (int i = tid; i < 16 * HS;  i += 512) sH1B[i] = (bf16_t)0.0f;
  for (int i = tid; i < 16 * AS2; i += 512) sA2[i]  = (bf16_t)0.0f;
  for (int i = tid; i < 16 * H_;  i += 512) { sC1[i] = 0.0f; sC2[i] = 0.0f; }

  // ---- x tile straight into registers, WMMA A layout ----------------------
  // A (16x32): lane<16 -> M=sub, K {0..7,16..23}; lane>=16 -> K {8..15,24..31}
  // All 16 waves fetch the same shared tile; lines stay hot in WGP$/L2.
  const float* xp = x + (r0 + sub) * (T_ * D_) + half * 8;
  v4f x0 = *(const v4f*)(xp +  0);
  v4f x1 = *(const v4f*)(xp +  4);
  v4f x2 = *(const v4f*)(xp + 16);
  v4f x3 = *(const v4f*)(xp + 20);

  // elementwise mapping: 512 threads, each does column ej of rows er, er+8
  const int ej = tid & 63;
  const int er = tid >> 6;

  wg_barrier();   // initial state visible

  for (int t = 0; t < T_; ++t) {
    // pack current x tile to bf16 A operand, then prefetch x_{t+1}
    v16bf ax;
    #pragma unroll
    for (int i = 0; i < 4; ++i) {
      ax[i]      = (bf16_t)x0[i];
      ax[4 + i]  = (bf16_t)x1[i];
      ax[8 + i]  = (bf16_t)x2[i];
      ax[12 + i] = (bf16_t)x3[i];
    }
    if (t + 1 < T_) {
      const float* xn = xp + (t + 1) * D_;
      x0 = *(const v4f*)(xn +  0);
      x1 = *(const v4f*)(xn +  4);
      x2 = *(const v4f*)(xn + 16);
      x3 = *(const v4f*)(xn + 20);
    }

    // ---- GEMM1: z1 = x@W1 + h1@U1 + b1  (K=96, 3 x wmma bf16) ----
    {
      v8f acc;
      #pragma unroll
      for (int i = 0; i < 8; ++i) acc[i] = bias1;
      acc = __builtin_amdgcn_wmma_f32_16x16x32_bf16(false, ax, false, b1r[0],
                                                    (short)0, acc, false, false);
      #pragma unroll
      for (int kt = 1; kt < 3; ++kt) {
        const bf16_t* p = &sH1B[sub * HS + (kt - 1) * 32 + half * 8];
        v8bf lo = *reinterpret_cast<const v8bf*>(p);        // ds_load_b128
        v8bf hi = *reinterpret_cast<const v8bf*>(p + 16);   // ds_load_b128
        v16bf a;
        #pragma unroll
        for (int i = 0; i < 8; ++i) { a[i] = lo[i]; a[8 + i] = hi[i]; }
        acc = __builtin_amdgcn_wmma_f32_16x16x32_bf16(false, a, false, b1r[kt],
                                                      (short)0, acc, false, false);
      }
      #pragma unroll
      for (int v = 0; v < 8; ++v)                 // C: lane=N, VGPR v -> M=half*8+v
        sZ1[(half * 8 + v) * ZS + nn] = acc[v];
    }
    wg_barrier();                                 // (A) z1 ready, h1 reads done

    // ---- layer-1 gates ----
    #pragma unroll
    for (int rr = 0; rr < 2; ++rr) {
      int r = er + rr * 8;
      float zi = sZ1[r * ZS + ej];
      float zf = sZ1[r * ZS + ej + 64];
      float zg = sZ1[r * ZS + ej + 128];
      float zo = sZ1[r * ZS + ej + 192];
      float ig = fsigmoid(zi);
      float fg = fsigmoid(zf);
      float gg = ftanh(zg);
      float og = fsigmoid(zo);
      float c  = fg * sC1[r * H_ + ej] + ig * gg;
      sC1[r * H_ + ej] = c;
      bf16_t hb = (bf16_t)(og * ftanh(c));
      sH1B[r * HS + ej]  = hb;   // h1 for next step's GEMM1
      sA2[r * AS2 + ej]  = hb;   // h1' for GEMM2 now
    }
    wg_barrier();                                 // (B) A2.h1' ready

    // ---- GEMM2: z2 = [h1' | h2] @ [W2;U2] + b2  (K=128, 4 x wmma bf16) ----
    {
      v8f acc;
      #pragma unroll
      for (int i = 0; i < 8; ++i) acc[i] = bias2;
      #pragma unroll
      for (int kt = 0; kt < 4; ++kt) {
        const bf16_t* p = &sA2[sub * AS2 + kt * 32 + half * 8];
        v8bf lo = *reinterpret_cast<const v8bf*>(p);
        v8bf hi = *reinterpret_cast<const v8bf*>(p + 16);
        v16bf a;
        #pragma unroll
        for (int i = 0; i < 8; ++i) { a[i] = lo[i]; a[8 + i] = hi[i]; }
        acc = __builtin_amdgcn_wmma_f32_16x16x32_bf16(false, a, false, b2r[kt],
                                                      (short)0, acc, false, false);
      }
      #pragma unroll
      for (int v = 0; v < 8; ++v)
        sZ2[(half * 8 + v) * ZS + nn] = acc[v];
    }
    wg_barrier();                                 // (C) z2 ready, A2 reads done

    // ---- layer-2 gates ----
    #pragma unroll
    for (int rr = 0; rr < 2; ++rr) {
      int r = er + rr * 8;
      float zi = sZ2[r * ZS + ej];
      float zf = sZ2[r * ZS + ej + 64];
      float zg = sZ2[r * ZS + ej + 128];
      float zo = sZ2[r * ZS + ej + 192];
      float ig = fsigmoid(zi);
      float fg = fsigmoid(zf);
      float gg = ftanh(zg);
      float og = fsigmoid(zo);
      float c  = fg * sC2[r * H_ + ej] + ig * gg;
      sC2[r * H_ + ej] = c;
      float h  = og * ftanh(c);
      sA2[r * AS2 + 64 + ej] = (bf16_t)h;   // h2 for next step
      sH2[r * H_ + ej] = h;                 // f32 copy for dense head
    }
    wg_barrier();                                 // (D) sH2 ready

    // ---- dense head: wave w reduces row w of h2 against Wd ----
    {
      float p = sH2[wave * H_ + lane] * wda + sH2[wave * H_ + lane + 32] * wdb;
      #pragma unroll
      for (int off = 16; off > 0; off >>= 1)
        p += __shfl_down(p, off, 32);
      if (lane == 0) {
        float o = fsigmoid(p + bd0);
        __builtin_nontemporal_store(o, &out[(r0 + wave) * T_ + t]);  // th:NT
      }
    }
  }
}

extern "C" void kernel_launch(void* const* d_in, const int* in_sizes, int n_in,
                              void* d_out, int out_size, void* d_ws, size_t ws_size,
                              hipStream_t stream) {
  (void)in_sizes; (void)n_in; (void)out_size; (void)d_ws; (void)ws_size;
  const float* x  = (const float*)d_in[0];
  const float* W1 = (const float*)d_in[1];
  const float* U1 = (const float*)d_in[2];
  const float* b1 = (const float*)d_in[3];
  const float* W2 = (const float*)d_in[4];
  const float* U2 = (const float*)d_in[5];
  const float* b2 = (const float*)d_in[6];
  const float* Wd = (const float*)d_in[7];
  const float* bd = (const float*)d_in[8];
  float* out = (float*)d_out;

  lstm2_fused_kernel<<<B_ / 16, 512, 0, stream>>>(x, W1, U1, b1, W2, U2, b2, Wd, bd, out);
}
